// Gemma4MoE_790273983122
// MI455X (gfx1250) — matrix operand order repaired and checked by reading, not compiled, not run on hardware
//
#include <hip/hip_runtime.h>
#include <hip/hip_bf16.h>
#include <cmath>

#define T_TOK 4096
#define HDIM  1024
#define NEXP  8
#define IDIM  1024

#define BM 32
#define BN 128
#define BK 32

typedef __attribute__((ext_vector_type(16))) __bf16 v16bf;
typedef __attribute__((ext_vector_type(8)))  float  v8f;

union BFrag { uint4 u[2]; v16bf v; };

__device__ __forceinline__ unsigned short f32_to_bf16_rne(float f) {
  unsigned u = __float_as_uint(f);
  unsigned r = u + 0x7FFFu + ((u >> 16) & 1u);
  return (unsigned short)(r >> 16);
}

__device__ __forceinline__ BFrag load_bfrag(const unsigned short* p) {
  BFrag b;
  b.u[0] = *(const uint4*)p;
  b.u[1] = *(const uint4*)(p + 8);
  return b;
}

// ---- CDNA5 async global->LDS staging (ASYNCcnt) with sync fallback --------
#if defined(__HIP_DEVICE_COMPILE__) && __has_builtin(__builtin_amdgcn_global_load_async_to_lds_b64)
#define HAVE_ASYNC_LDS 1
typedef int v2i_t __attribute__((ext_vector_type(2)));
typedef __attribute__((address_space(1))) v2i_t* gv2i;
typedef __attribute__((address_space(3))) v2i_t* lv2i;
#else
#define HAVE_ASYNC_LDS 0
#endif

__device__ __forceinline__ void stage_b64(const unsigned short* g, unsigned short* l) {
#if HAVE_ASYNC_LDS
  __builtin_amdgcn_global_load_async_to_lds_b64((gv2i)(void*)g, (lv2i)(void*)l, 0, 0);
#else
  *(uint2*)l = *(const uint2*)g;
#endif
}

// Wait for this wave's LDS staging to be complete (before barrier signal).
__device__ __forceinline__ void stage_wait() {
#if HAVE_ASYNC_LDS
#if __has_builtin(__builtin_amdgcn_s_wait_asynccnt)
  __builtin_amdgcn_s_wait_asynccnt(0);
#else
  asm volatile("s_wait_asynccnt 0x0" ::: "memory");
#endif
#else
  asm volatile("s_wait_dscnt 0x0" ::: "memory");
#endif
}

// Raw split workgroup barrier WITHOUT the full acq/rel VMEM drain that
// __syncthreads() implies (which forces s_wait_loadcnt 0 on in-flight,
// VGPR-destined B loads). LDS ordering is guaranteed by stage_wait()
// before the signal; asm memory clobbers stop compiler reordering.
__device__ __forceinline__ void wg_barrier() {
#if defined(__HIP_DEVICE_COMPILE__) && __has_builtin(__builtin_amdgcn_s_barrier_signal) && \
    __has_builtin(__builtin_amdgcn_s_barrier_wait)
  asm volatile("" ::: "memory");
  __builtin_amdgcn_s_barrier_signal(-1);
  __builtin_amdgcn_s_barrier_wait(-1);
  asm volatile("" ::: "memory");
#else
  __syncthreads();
#endif
}

// ---------------------------------------------------------------------------
// Router: softmax over E=8 logits, top-2, renormalized combine weights,
// atomic append of (token, weight) into per-expert lists.
// ---------------------------------------------------------------------------
__global__ void router_kernel(const float* __restrict__ logits,
                              const float* __restrict__ scale,
                              int* __restrict__ counts,
                              int* __restrict__ tids,
                              float* __restrict__ tw) {
  int t = blockIdx.x * blockDim.x + threadIdx.x;
  if (t >= T_TOK) return;
  float l[NEXP];
#pragma unroll
  for (int e = 0; e < NEXP; ++e) l[e] = logits[t * NEXP + e];
  float mx = l[0];
#pragma unroll
  for (int e = 1; e < NEXP; ++e) mx = fmaxf(mx, l[e]);
  float p[NEXP];
#pragma unroll
  for (int e = 0; e < NEXP; ++e) p[e] = __expf(l[e] - mx);
  int i1 = 0;
#pragma unroll
  for (int e = 1; e < NEXP; ++e) if (l[e] > l[i1]) i1 = e;
  int i2 = (i1 == 0) ? 1 : 0;
#pragma unroll
  for (int e = 0; e < NEXP; ++e) if (e != i1 && l[e] > l[i2]) i2 = e;
  float denom = p[i1] + p[i2];
  float wa = p[i1] / denom * scale[i1];
  float wb = p[i2] / denom * scale[i2];
  int s1 = atomicAdd(&counts[i1], 1);
  tids[i1 * T_TOK + s1] = t;
  tw[i1 * T_TOK + s1] = wa;
  int s2 = atomicAdd(&counts[i2], 1);
  tids[i2 * T_TOK + s2] = t;
  tw[i2 * T_TOK + s2] = wb;
}

// ---------------------------------------------------------------------------
__global__ void cvt_f32_bf16_kernel(const float* __restrict__ src,
                                    unsigned short* __restrict__ dst, int n4) {
  int i = blockIdx.x * blockDim.x + threadIdx.x;
  int stride = gridDim.x * blockDim.x;
  for (; i < n4; i += stride) {
    float4 f = reinterpret_cast<const float4*>(src)[i];
    ushort4 o;
    o.x = f32_to_bf16_rne(f.x);
    o.y = f32_to_bf16_rne(f.y);
    o.z = f32_to_bf16_rne(f.z);
    o.w = f32_to_bf16_rne(f.w);
    reinterpret_cast<ushort4*>(dst)[i] = o;
  }
}

__global__ void zero_f32_kernel(float* __restrict__ p, int n) {
  int i = blockIdx.x * blockDim.x + threadIdx.x;
  int stride = gridDim.x * blockDim.x;
  for (; i < n; i += stride) p[i] = 0.f;
}

// ---------------------------------------------------------------------------
// GEMM1: gathered xb rows x (w1^T, w3^T); fused exact-GeLU(g)*u -> hbuf bf16.
// 256 threads = 8 waves (2 m x 4 n). Double-buffered LDS A tile staged with
// async global->LDS; register-pipelined B fragments (next K-step's loads
// issue before current WMMAs).
// ---------------------------------------------------------------------------
__global__ __launch_bounds__(256) void moe_gemm1_kernel(
    const unsigned short* __restrict__ xb,
    const unsigned short* __restrict__ w13b,
    const int* __restrict__ counts,
    const int* __restrict__ tids,
    unsigned short* __restrict__ hbuf) {
  const int e = blockIdx.x;
  const int m0 = blockIdx.y * BM;
  const int n0 = blockIdx.z * BN;
  const int cnt = counts[e];
  if (m0 >= cnt) return;  // uniform per block

  __shared__ unsigned short As[2][BM][BK];
  __shared__ int rowids[BM];

  const int tid = threadIdx.x;
  const int lane = tid & 31;
  const int wave = tid >> 5;
  const int mw = wave >> 2;  // 0..1
  const int nw = wave & 3;   // 0..3

  if (tid < BM) {
    int slot = m0 + tid;
    rowids[tid] = tids[e * T_TOK + ((slot < cnt) ? slot : (cnt - 1))];
  }
  __syncthreads();

  const unsigned short* w1base = w13b + (size_t)e * (2 * IDIM) * HDIM;
  const unsigned short* w3base = w1base + (size_t)IDIM * HDIM;

  const int arow = tid >> 3;       // LDS tile row this thread stages
  const int acol = (tid & 7) * 4;  // 4 bf16 (8B) per thread
  const int mr  = mw * 16 + (lane & 15);
  const int kbA = (lane < 16) ? 0 : 16;
  const int kbB = (lane < 16) ? 0 : 16;
  const int nA  = n0 + nw * 32 + (lane & 15);
  const int nB  = nA + 16;

  // Loop-invariant source/dest addresses.
  const unsigned short* a_src = xb + (size_t)rowids[arow] * HDIM + acol;
  unsigned short* a_dst0 = &As[0][arow][acol];
  unsigned short* a_dst1 = &As[1][arow][acol];
  const unsigned short* pg0 = w1base + (size_t)nA * HDIM + kbB;
  const unsigned short* pg1 = w1base + (size_t)nB * HDIM + kbB;
  const unsigned short* pu0 = w3base + (size_t)nA * HDIM + kbB;
  const unsigned short* pu1 = w3base + (size_t)nB * HDIM + kbB;
  const unsigned short* aF0 = &As[0][mr][kbA];
  const unsigned short* aF1 = &As[1][mr][kbA];

  v8f ag0 = {}, ag1 = {}, au0 = {}, au1 = {};

  // Prologue: stage A tile 0, load B fragments for k=0.
  stage_b64(a_src, a_dst0);
  BFrag bg0 = load_bfrag(pg0);
  BFrag bg1 = load_bfrag(pg1);
  BFrag bu0 = load_bfrag(pu0);
  BFrag bu1 = load_bfrag(pu1);
  stage_wait();
  wg_barrier();

  int buf = 0;
#pragma unroll 2
  for (int kk = 0; kk < HDIM; kk += BK) {
    const int knext = kk + BK;
    const bool has_next = knext < HDIM;
    // Stage next A tile into the other LDS buffer (overlaps WMMAs below).
    if (has_next) stage_b64(a_src + knext, buf ? a_dst0 : a_dst1);
    // Issue next K-step's B loads before consuming current fragments.
    const int kl = has_next ? knext : 0;
    BFrag ng0 = load_bfrag(pg0 + kl);
    BFrag ng1 = load_bfrag(pg1 + kl);
    BFrag nu0 = load_bfrag(pu0 + kl);
    BFrag nu1 = load_bfrag(pu1 + kl);

    // Current A fragment from resident LDS buffer.
    const unsigned short* af = buf ? aF1 : aF0;
    BFrag a;
    a.u[0] = *(const uint4*)af;
    a.u[1] = *(const uint4*)(af + 8);

    ag0 = __builtin_amdgcn_wmma_f32_16x16x32_bf16(false, a.v, false, bg0.v,
                                                  (short)0, ag0, false, false);
    ag1 = __builtin_amdgcn_wmma_f32_16x16x32_bf16(false, a.v, false, bg1.v,
                                                  (short)0, ag1, false, false);
    au0 = __builtin_amdgcn_wmma_f32_16x16x32_bf16(false, a.v, false, bu0.v,
                                                  (short)0, au0, false, false);
    au1 = __builtin_amdgcn_wmma_f32_16x16x32_bf16(false, a.v, false, bu1.v,
                                                  (short)0, au1, false, false);

    bg0 = ng0; bg1 = ng1; bu0 = nu0; bu1 = nu1;
    stage_wait();
    wg_barrier();
    buf ^= 1;
  }

  // Epilogue: h = gelu_exact(g)*u, bf16 store.
  const int mbase = m0 + mw * 16 + ((lane < 16) ? 0 : 8);
#pragma unroll
  for (int r = 0; r < 8; ++r) {
    size_t off = ((size_t)e * T_TOK + (mbase + r)) * IDIM;
    float g0 = ag0[r];
    float h0 = 0.5f * g0 * (1.f + erff(g0 * 0.70710678118f)) * au0[r];
    hbuf[off + nA] = f32_to_bf16_rne(h0);
    float g1 = ag1[r];
    float h1 = 0.5f * g1 * (1.f + erff(g1 * 0.70710678118f)) * au1[r];
    hbuf[off + nB] = f32_to_bf16_rne(h1);
  }
}

// ---------------------------------------------------------------------------
// GEMM2: y = h @ w2^T; epilogue scatters weight*y into out with f32 atomics.
// ---------------------------------------------------------------------------
__global__ __launch_bounds__(256) void moe_gemm2_kernel(
    const unsigned short* __restrict__ hbuf,
    const unsigned short* __restrict__ w2b,
    const int* __restrict__ counts,
    const int* __restrict__ tids,
    const float* __restrict__ tw,
    float* __restrict__ out) {
  const int e = blockIdx.x;
  const int m0 = blockIdx.y * BM;
  const int n0 = blockIdx.z * BN;
  const int cnt = counts[e];
  if (m0 >= cnt) return;

  __shared__ unsigned short As[2][BM][BK];
  __shared__ int rowt[BM];
  __shared__ float roww[BM];

  const int tid = threadIdx.x;
  const int lane = tid & 31;
  const int wave = tid >> 5;
  const int mw = wave >> 2;
  const int nw = wave & 3;

  if (tid < BM) {
    int slot = m0 + tid;
    bool valid = slot < cnt;
    rowt[tid] = valid ? tids[e * T_TOK + slot] : -1;
    roww[tid] = valid ? tw[e * T_TOK + slot] : 0.f;
  }
  __syncthreads();

  const unsigned short* w2base = w2b + (size_t)e * HDIM * IDIM;

  const int arow = tid >> 3;
  const int acol = (tid & 7) * 4;
  const int mr  = mw * 16 + (lane & 15);
  const int kbA = (lane < 16) ? 0 : 16;
  const int kbB = (lane < 16) ? 0 : 16;
  const int nA  = n0 + nw * 32 + (lane & 15);
  const int nB  = nA + 16;

  const unsigned short* a_src =
      hbuf + ((size_t)e * T_TOK + m0 + arow) * IDIM + acol;
  unsigned short* a_dst0 = &As[0][arow][acol];
  unsigned short* a_dst1 = &As[1][arow][acol];
  const unsigned short* p0 = w2base + (size_t)nA * IDIM + kbB;
  const unsigned short* p1 = w2base + (size_t)nB * IDIM + kbB;
  const unsigned short* aF0 = &As[0][mr][kbA];
  const unsigned short* aF1 = &As[1][mr][kbA];

  v8f c0 = {}, c1 = {};

  stage_b64(a_src, a_dst0);
  BFrag b0 = load_bfrag(p0);
  BFrag b1 = load_bfrag(p1);
  stage_wait();
  wg_barrier();

  int buf = 0;
#pragma unroll 2
  for (int kk = 0; kk < IDIM; kk += BK) {
    const int knext = kk + BK;
    const bool has_next = knext < IDIM;
    if (has_next) stage_b64(a_src + knext, buf ? a_dst0 : a_dst1);
    const int kl = has_next ? knext : 0;
    BFrag n0f = load_bfrag(p0 + kl);
    BFrag n1f = load_bfrag(p1 + kl);

    const unsigned short* af = buf ? aF1 : aF0;
    BFrag a;
    a.u[0] = *(const uint4*)af;
    a.u[1] = *(const uint4*)(af + 8);

    c0 = __builtin_amdgcn_wmma_f32_16x16x32_bf16(false, a.v, false, b0.v,
                                                 (short)0, c0, false, false);
    c1 = __builtin_amdgcn_wmma_f32_16x16x32_bf16(false, a.v, false, b1.v,
                                                 (short)0, c1, false, false);

    b0 = n0f; b1 = n1f;
    stage_wait();
    wg_barrier();
    buf ^= 1;
  }

  const int mloc = mw * 16 + ((lane < 16) ? 0 : 8);
#pragma unroll
  for (int r = 0; r < 8; ++r) {
    int slot = mloc + r;
    int t = rowt[slot];
    if (t >= 0) {
      float w = roww[slot];
      atomicAdd(&out[(size_t)t * HDIM + nA], w * c0[r]);
      atomicAdd(&out[(size_t)t * HDIM + nB], w * c1[r]);
    }
  }
}

// ---------------------------------------------------------------------------
extern "C" void kernel_launch(void* const* d_in, const int* in_sizes, int n_in,
                              void* d_out, int out_size, void* d_ws, size_t ws_size,
                              hipStream_t stream) {
  (void)in_sizes; (void)n_in; (void)out_size; (void)ws_size;
  const float* x   = (const float*)d_in[0];
  const float* rl  = (const float*)d_in[1];
  const float* pes = (const float*)d_in[2];
  const float* w13 = (const float*)d_in[3];
  const float* w2  = (const float*)d_in[4];
  float* out = (float*)d_out;

  char* ws = (char*)d_ws;
  size_t off = 0;
  auto alloc = [&](size_t bytes) -> void* {
    size_t a = (off + 255) & ~(size_t)255;
    off = a + bytes;
    return (void*)(ws + a);
  };
  int*   counts = (int*)alloc(NEXP * sizeof(int));
  int*   tids   = (int*)alloc((size_t)NEXP * T_TOK * sizeof(int));
  float* tw     = (float*)alloc((size_t)NEXP * T_TOK * sizeof(float));
  unsigned short* xb   = (unsigned short*)alloc((size_t)T_TOK * HDIM * 2);
  unsigned short* w13b = (unsigned short*)alloc((size_t)NEXP * 2 * IDIM * HDIM * 2);
  unsigned short* w2b  = (unsigned short*)alloc((size_t)NEXP * HDIM * IDIM * 2);
  unsigned short* hbuf = (unsigned short*)alloc((size_t)NEXP * T_TOK * IDIM * 2);

  zero_f32_kernel<<<1, 64, 0, stream>>>((float*)counts, NEXP);
  zero_f32_kernel<<<dim3(2048), 256, 0, stream>>>(out, T_TOK * HDIM);
  router_kernel<<<dim3((T_TOK + 255) / 256), 256, 0, stream>>>(rl, pes, counts, tids, tw);
  cvt_f32_bf16_kernel<<<dim3(2048), 256, 0, stream>>>(x, xb, T_TOK * HDIM / 4);
  cvt_f32_bf16_kernel<<<dim3(4096), 256, 0, stream>>>(w13, w13b, NEXP * 2 * IDIM * HDIM / 4);
  cvt_f32_bf16_kernel<<<dim3(4096), 256, 0, stream>>>(w2, w2b, NEXP * HDIM * IDIM / 4);
  moe_gemm1_kernel<<<dim3(NEXP, T_TOK / BM, IDIM / BN), 256, 0, stream>>>(
      xb, w13b, counts, tids, hbuf);
  moe_gemm2_kernel<<<dim3(NEXP, T_TOK / BM, HDIM / BN), 256, 0, stream>>>(
      hbuf, w2b, counts, tids, tw, out);
}
